// CNNMambaFast_78932908966461
// MI455X (gfx1250) — matrix operand order repaired and checked by reading, not compiled, not run on hardware
//
#include <hip/hip_runtime.h>
#include <hip/hip_bf16.h>

// ---------------------------------------------------------------------------
// CNN + Mamba (4 layers) + head, for MI455X (gfx1250, wave32, WMMA).
// GEMMs: async global->LDS double-buffered staging (ASYNCcnt, SADDR form) +
// XOR-swizzled LDS fragment reads (ds_load_b128) + v_wmma_f32_16x16x32_bf16.
// Block tile 128Mx64N (4 waves, each 32Mx64N => 8 WMMA/stage/wave).
// Scan parallelized over (B x DI) with Bt/Ct staged in LDS.
// ---------------------------------------------------------------------------

typedef __bf16 bf16_t;
typedef __attribute__((ext_vector_type(16))) __bf16 v16bf;
typedef __attribute__((ext_vector_type(8)))  __bf16 v8bf;
typedef __attribute__((ext_vector_type(8)))  float   v8f;

#define B_   4
#define M_   128
#define T_   1024
#define NL_  4
#define DM_  512
#define DI_  1024
#define NST_ 16
#define DC_  4
#define DR_  32
#define NC_  10
#define ND_  256
#define TOK  (B_ * T_)      // 4096 tokens
#define HN2  64             // padded head2 out dim

#define STAGE_BYTES 12288   // A tile 128x32 bf16 (8KB) + W tile 64x32 bf16 (4KB)
#define WOFF        8192    // W tile offset within a stage

__device__ __forceinline__ float silu_f(float v) { return v / (1.f + __expf(-v)); }
__device__ __forceinline__ float softplus_f(float v) {
    return (v > 20.f) ? v : log1pf(__expf(v));
}

// --------------------------- generic WMMA GEMM -----------------------------
// out = act(A[M,K](bf16, row stride lda) @ W[N,K]^T(bf16) + bias) (+resid)
// act: 0 none, 1 silu, 2 softplus. outf (f32) and/or outb (bf16), stride N.
// Requires: M % 128 == 0, N % 64 == 0, K % 32 == 0, rows 16B-aligned.
__global__ void __launch_bounds__(128)
gemm_bf16_wmma(const bf16_t* __restrict__ A,
               const bf16_t* __restrict__ W,
               const float*  __restrict__ bias,
               const float*  __restrict__ resid,
               float*        __restrict__ outf,
               bf16_t*       __restrict__ outb,
               int M, int N, int K, int lda, int act) {
    __shared__ __align__(16) char smem[2 * STAGE_BYTES];
    const int tid    = threadIdx.x;
    const int lane   = tid & 31;
    const int wave   = tid >> 5;
    const int mblock = blockIdx.y * 128;
    const int nblock = blockIdx.x * 64;
    const unsigned smbase = (unsigned)(unsigned long long)(&smem[0]);

    // Per-thread, loop-invariant 32-bit byte offsets (SADDR holds base + kk).
    // A tile: 512 16B chunks (rows 0..127); W tile: 256 chunks (rows 0..63).
    unsigned voffA[4], loffA[4], voffW[2], loffW[2];
    #pragma unroll
    for (int j = 0; j < 4; ++j) {
        int lin = j * 128 + tid;
        int row = lin >> 2;
        int ch  = lin & 3;
        voffA[j] = (unsigned)((row * lda + ch * 8) * 2);
        loffA[j] = smbase + (unsigned)(row * 64) +
                   (unsigned)(((ch ^ ((row >> 2) & 3)) * 16));
    }
    #pragma unroll
    for (int j = 0; j < 2; ++j) {
        int lin = j * 128 + tid;
        int row = lin >> 2;
        int ch  = lin & 3;
        voffW[j] = (unsigned)((row * K + ch * 8) * 2);
        loffW[j] = smbase + WOFF + (unsigned)(row * 64) +
                   (unsigned)(((ch ^ ((row >> 2) & 3)) * 16));
    }

    // 6 async b128 copies per thread per stage; scalar base advances with kk.
    auto issue_stage = [&](int kk, int buf) {
        unsigned bo = (unsigned)buf * (unsigned)STAGE_BYTES;
        unsigned long long baseA =
            (unsigned long long)(A + (long)mblock * lda + kk);
        unsigned long long baseW =
            (unsigned long long)(W + (long)nblock * K + kk);
        #pragma unroll
        for (int j = 0; j < 4; ++j)
            asm volatile("global_load_async_to_lds_b128 %0, %1, %2"
                         :: "v"(loffA[j] + bo), "v"(voffA[j]), "s"(baseA)
                         : "memory");
        #pragma unroll
        for (int j = 0; j < 2; ++j)
            asm volatile("global_load_async_to_lds_b128 %0, %1, %2"
                         :: "v"(loffW[j] + bo), "v"(voffW[j]), "s"(baseW)
                         : "memory");
    };

    v8f acc[2][4] = {};
    // wave tile: rows [wave*32, wave*32+32), all 64 cols of the block tile.
    // One B fragment live at a time to keep register pressure low.
    auto compute_stage = [&](int buf) {
        const int bufo = buf * STAGE_BYTES;
        const int c0 = (lane & 16) ? 1 : 0;           // A chunks {c0, c0+2}
        v16bf a[2];
        #pragma unroll
        for (int mi = 0; mi < 2; ++mi) {
            int r  = wave * 32 + mi * 16 + (lane & 15);
            int sw = (r >> 2) & 3;
            union { v16bf v; v8bf h[2]; } u;
            u.h[0] = *(const v8bf*)&smem[bufo + r * 64 + ((c0       ^ sw) * 16)];
            u.h[1] = *(const v8bf*)&smem[bufo + r * 64 + (((c0 + 2) ^ sw) * 16)];
            a[mi] = u.v;
        }
        const int c0b = (lane & 16) ? 1 : 0;          // B chunks {c0b, c0b+2}
        #pragma unroll
        for (int ni = 0; ni < 4; ++ni) {
            int r  = ni * 16 + (lane & 15);
            int sw = (r >> 2) & 3;
            union { v16bf v; v8bf h[2]; } u;
            u.h[0] = *(const v8bf*)&smem[bufo + WOFF + r * 64 + ((c0b       ^ sw) * 16)];
            u.h[1] = *(const v8bf*)&smem[bufo + WOFF + r * 64 + (((c0b + 2) ^ sw) * 16)];
            acc[0][ni] = __builtin_amdgcn_wmma_f32_16x16x32_bf16(
                false, a[0], false, u.v, (short)0, acc[0][ni], false, false);
            acc[1][ni] = __builtin_amdgcn_wmma_f32_16x16x32_bf16(
                false, a[1], false, u.v, (short)0, acc[1][ni], false, false);
        }
    };

    const int nst = K / 32;
    issue_stage(0, 0);
    int s = 0;
    for (; s < nst - 1; ++s) {                        // steady state (no branches)
        issue_stage((s + 1) * 32, (s + 1) & 1);
        asm volatile("s_wait_asynccnt 0x6" ::: "memory");   // stage s landed
        __syncthreads();
        compute_stage(s & 1);
        __syncthreads();                              // buffer free for stage s+2
    }
    asm volatile("s_wait_asynccnt 0x0" ::: "memory");
    __syncthreads();
    compute_stage(s & 1);

    // C/D layout: lane<16 -> rows M=e, N=lane; lane>=16 -> rows M=8+e, N=lane-16
    const int mbase  = mblock + wave * 32;
    const int ncol0  = nblock + (lane & 15);
    const int rowoff = (lane & 16) ? 8 : 0;
    #pragma unroll
    for (int mi = 0; mi < 2; ++mi) {
        #pragma unroll
        for (int ni = 0; ni < 4; ++ni) {
            v8f av = acc[mi][ni];
            int col = ncol0 + ni * 16;
            float bv = bias ? bias[col] : 0.f;
            #pragma unroll
            for (int e = 0; e < 8; ++e) {
                int row = mbase + mi * 16 + rowoff + e;
                float v = av[e] + bv;
                if (act == 1)      v = silu_f(v);
                else if (act == 2) v = softplus_f(v);
                long idx = (long)row * N + col;
                if (resid) v += resid[idx];
                if (outf)  outf[idx] = v;
                if (outb)  outb[idx] = (bf16_t)v;
            }
        }
    }
}

// ------------------------------ elementwise --------------------------------
__global__ void cvt_f32_bf16(const float* __restrict__ s, bf16_t* __restrict__ d, long n) {
    long i = (long)blockIdx.x * 256 + threadIdx.x;
    if (i < n) d[i] = (bf16_t)s[i];
}

// feat[(b*T+t)*256 + c] : c<128 -> x[b,c,t] ; else relu(x[b,c-128,t]-x[b,c-128,t-1])
__global__ void feat_prep(const float* __restrict__ x, bf16_t* __restrict__ feat) {
    long idx = (long)blockIdx.x * 256 + threadIdx.x;   // TOK*ND
    int c = idx & 255;
    long token = idx >> 8;
    int b = (int)(token >> 10), t = (int)(token & 1023);
    float v;
    if (c < 128) {
        v = x[((long)b * M_ + c) * T_ + t];
    } else {
        int ch = c - 128;
        if (t == 0) v = 0.f;
        else {
            float cur  = x[((long)b * M_ + ch) * T_ + t];
            float prev = x[((long)b * M_ + ch) * T_ + t - 1];
            v = fmaxf(cur - prev, 0.f);
        }
    }
    feat[idx] = (bf16_t)v;
}

// LayerNorm over D=512 per token; writes bf16. One block (128 thr) per token.
__global__ void layernorm_bf16(const float* __restrict__ h,
                               const float* __restrict__ g,
                               const float* __restrict__ bta,
                               bf16_t* __restrict__ out, int D) {
    int token = blockIdx.x;
    const float* row = h + (long)token * D;
    float s = 0.f, s2 = 0.f;
    for (int c = threadIdx.x; c < D; c += 128) { float v = row[c]; s += v; s2 += v * v; }
    __shared__ float r0[128], r1[128];
    r0[threadIdx.x] = s; r1[threadIdx.x] = s2;
    __syncthreads();
    for (int off = 64; off > 0; off >>= 1) {
        if (threadIdx.x < off) {
            r0[threadIdx.x] += r0[threadIdx.x + off];
            r1[threadIdx.x] += r1[threadIdx.x + off];
        }
        __syncthreads();
    }
    float mean = r0[0] / D;
    float var  = r1[0] / D - mean * mean;
    float rstd = rsqrtf(var + 1e-5f);
    for (int c = threadIdx.x; c < D; c += 128)
        out[(long)token * D + c] = (bf16_t)((row[c] - mean) * rstd * g[c] + bta[c]);
}

// depthwise causal conv (DC=4) over u = xz[:, 0:1024] + silu; dual f32/bf16 out
__global__ void conv_silu(const float* __restrict__ xz,
                          const float* __restrict__ cw,   // [DI,DC]
                          const float* __restrict__ cb,   // [DI]
                          float* __restrict__ ucf, bf16_t* __restrict__ ucb) {
    long idx = (long)blockIdx.x * 256 + threadIdx.x;      // TOK*DI
    int d = idx & (DI_ - 1);
    long token = idx >> 10;
    int b = (int)(token >> 10), t = (int)(token & 1023);
    float acc = cb[d];
    #pragma unroll
    for (int k = 0; k < DC_; ++k) {
        int tt = t - (DC_ - 1) + k;
        if (tt >= 0)
            acc += xz[(((long)b << 10) + tt) * (2 * DI_) + d] * cw[d * DC_ + k];
    }
    float v = silu_f(acc);
    ucf[idx] = v;
    ucb[idx] = (bf16_t)v;
}

// selective scan: one thread per (b,d); Bt/Ct staged in LDS per step.
__global__ void mamba_scan(const float* __restrict__ dbl,    // [TOK,64] dt|B|C
                           const float* __restrict__ delta,  // [TOK,DI]
                           const float* __restrict__ ucf,    // [TOK,DI]
                           const float* __restrict__ Alog,   // [DI,NST]
                           float* __restrict__ ys) {         // [TOK,DI]
    int b = blockIdx.y;
    int d = blockIdx.x * 256 + threadIdx.x;
    float An[NST_];
    #pragma unroll
    for (int n = 0; n < NST_; ++n) An[n] = -__expf(Alog[(long)d * NST_ + n]);
    float st[NST_];
    #pragma unroll
    for (int n = 0; n < NST_; ++n) st[n] = 0.f;

    __shared__ float sBC[32];   // [0..15]=Bt, [16..31]=Ct
    for (int t = 0; t < T_; ++t) {
        long token = ((long)b << 10) + t;
        if (threadIdx.x < 32) sBC[threadIdx.x] = dbl[token * 64 + 32 + threadIdx.x];
        __syncthreads();
        float dlt = delta[token * DI_ + d];
        float du  = dlt * ucf[token * DI_ + d];
        float y = 0.f;
        #pragma unroll
        for (int n = 0; n < NST_; ++n) {
            st[n] = __expf(dlt * An[n]) * st[n] + du * sBC[n];
            y += st[n] * sBC[16 + n];
        }
        ys[token * DI_ + d] = y;
        __syncthreads();
    }
}

// y = (ys + uc*D_skip) * silu(z),  z = xz[:,1024:2048]; bf16 out
__global__ void gate_out(const float* __restrict__ xz,
                         const float* __restrict__ ys,
                         const float* __restrict__ ucf,
                         const float* __restrict__ Dskip,
                         bf16_t* __restrict__ yb) {
    long idx = (long)blockIdx.x * 256 + threadIdx.x;      // TOK*DI
    int d = idx & (DI_ - 1);
    long token = idx >> 10;
    float z = xz[token * (2 * DI_) + DI_ + d];
    float v = (ys[idx] + ucf[idx] * Dskip[d]) * silu_f(z);
    yb[idx] = (bf16_t)v;
}

// pad head2 weights (10,128)->(64,128) bf16, bias (10)->(64) f32
__global__ void pad_head2(const float* __restrict__ w, const float* __restrict__ b,
                          bf16_t* __restrict__ wp, float* __restrict__ bp) {
    int idx = blockIdx.x * 256 + threadIdx.x;             // 64*128
    int r = idx >> 7;
    wp[idx] = (bf16_t)((r < NC_) ? w[r * 128 + (idx & 127)] : 0.f);
    if (idx < HN2) bp[idx] = (idx < NC_) ? b[idx] : 0.f;
}

// out[b, c, t] = logits[(b*T+t)*HN2 + c]
__global__ void write_out(const float* __restrict__ logits, float* __restrict__ out) {
    int idx = blockIdx.x * 256 + threadIdx.x;             // B*NC*T = 40960
    int t = idx & 1023;
    int rest = idx >> 10;
    int c = rest % NC_;
    int b = rest / NC_;
    out[idx] = logits[(((long)b << 10) + t) * HN2 + c];
}

// ---------------------------------------------------------------------------
extern "C" void kernel_launch(void* const* d_in, const int* in_sizes, int n_in,
                              void* d_out, int out_size, void* d_ws, size_t ws_size,
                              hipStream_t stream) {
    const float* x      = (const float*)d_in[0];
    const float* fc1_w  = (const float*)d_in[1];
    const float* fc1_b  = (const float*)d_in[2];
    const float* fc2_w  = (const float*)d_in[3];
    const float* fc2_b  = (const float*)d_in[4];
    const float* proj_w = (const float*)d_in[5];
    const float* proj_b = (const float*)d_in[6];
    const float* norm_g = (const float*)d_in[7];
    const float* norm_b = (const float*)d_in[8];
    const float* in_w   = (const float*)d_in[9];
    const float* conv_w = (const float*)d_in[10];
    const float* conv_b = (const float*)d_in[11];
    const float* xproj_w= (const float*)d_in[12];
    const float* dt_w   = (const float*)d_in[13];
    const float* dt_b   = (const float*)d_in[14];
    const float* A_log  = (const float*)d_in[15];
    const float* D_skip = (const float*)d_in[16];
    const float* out_w  = (const float*)d_in[17];
    const float* hn_g   = (const float*)d_in[18];
    const float* hn_b   = (const float*)d_in[19];
    const float* h1_w   = (const float*)d_in[20];
    const float* h1_b   = (const float*)d_in[21];
    const float* h2_w   = (const float*)d_in[22];
    const float* h2_b   = (const float*)d_in[23];

    char* base = (char*)d_ws;
    size_t off = 0;
    auto alloc = [&](size_t bytes) -> void* {
        off = (off + 255) & ~(size_t)255;
        void* p = base + off;
        off += bytes;
        return p;
    };

    // bf16 weights
    bf16_t* w_fc1  = (bf16_t*)alloc((size_t)512 * 256 * 2);
    bf16_t* w_fc2  = (bf16_t*)alloc((size_t)256 * 512 * 2);
    bf16_t* w_proj = (bf16_t*)alloc((size_t)512 * 256 * 2);
    bf16_t* w_in   = (bf16_t*)alloc((size_t)NL_ * 2048 * 512 * 2);
    bf16_t* w_xprj = (bf16_t*)alloc((size_t)NL_ * 64 * 1024 * 2);
    bf16_t* w_dt   = (bf16_t*)alloc((size_t)NL_ * 1024 * 32 * 2);
    bf16_t* w_out  = (bf16_t*)alloc((size_t)NL_ * 512 * 1024 * 2);
    bf16_t* w_h1   = (bf16_t*)alloc((size_t)128 * 512 * 2);
    bf16_t* w_h2p  = (bf16_t*)alloc((size_t)HN2 * 128 * 2);
    float*  b_h2p  = (float*) alloc((size_t)HN2 * 4);
    // activations
    bf16_t* featb  = (bf16_t*)alloc((size_t)TOK * ND_ * 2);
    bf16_t* h1b    = (bf16_t*)alloc((size_t)TOK * 512 * 2);
    bf16_t* h2b    = (bf16_t*)alloc((size_t)TOK * 256 * 2);
    float*  hres   = (float*) alloc((size_t)TOK * DM_ * 4);
    bf16_t* hnb    = (bf16_t*)alloc((size_t)TOK * DM_ * 2);
    float*  xz     = (float*) alloc((size_t)TOK * 2 * DI_ * 4);
    float*  ucf    = (float*) alloc((size_t)TOK * DI_ * 4);
    bf16_t* ucb    = (bf16_t*)alloc((size_t)TOK * DI_ * 2);
    float*  dbl    = (float*) alloc((size_t)TOK * 64 * 4);
    bf16_t* dblb   = (bf16_t*)alloc((size_t)TOK * 64 * 2);
    float*  delta  = (float*) alloc((size_t)TOK * DI_ * 4);
    float*  ys     = (float*) alloc((size_t)TOK * DI_ * 4);
    bf16_t* yb     = (bf16_t*)alloc((size_t)TOK * DI_ * 2);
    bf16_t* t1b    = (bf16_t*)alloc((size_t)TOK * 128 * 2);
    float*  logits = (float*) alloc((size_t)TOK * HN2 * 4);

    auto cvt = [&](const float* s, bf16_t* d, long n) {
        cvt_f32_bf16<<<dim3((unsigned)((n + 255) / 256)), dim3(256), 0, stream>>>(s, d, n);
    };
    auto gemm = [&](const bf16_t* A, const bf16_t* W, const float* bias, const float* resid,
                    float* outf, bf16_t* outb, int M, int N, int K, int lda, int act) {
        gemm_bf16_wmma<<<dim3(N / 64, M / 128), dim3(128), 0, stream>>>(
            A, W, bias, resid, outf, outb, M, N, K, lda, act);
    };

    // weight conversion (deterministic per launch)
    cvt(fc1_w,  w_fc1,  512 * 256);
    cvt(fc2_w,  w_fc2,  256 * 512);
    cvt(proj_w, w_proj, 512 * 256);
    cvt(in_w,   w_in,   (long)NL_ * 2048 * 512);
    cvt(xproj_w,w_xprj, (long)NL_ * 64 * 1024);
    cvt(dt_w,   w_dt,   (long)NL_ * 1024 * 32);
    cvt(out_w,  w_out,  (long)NL_ * 512 * 1024);
    cvt(h1_w,   w_h1,   128 * 512);
    pad_head2<<<dim3(32), dim3(256), 0, stream>>>(h2_w, h2_b, w_h2p, b_h2p);

    // front end
    feat_prep<<<dim3(TOK * ND_ / 256), dim3(256), 0, stream>>>(x, featb);
    gemm(featb, w_fc1, fc1_b, nullptr, nullptr, h1b, TOK, 512, 256, 256, 1);  // silu
    gemm(h1b,   w_fc2, fc2_b, nullptr, nullptr, h2b, TOK, 256, 512, 512, 0);
    gemm(h2b,   w_proj, proj_b, nullptr, hres, nullptr, TOK, 512, 256, 256, 0);

    // mamba layers
    for (int i = 0; i < NL_; ++i) {
        layernorm_bf16<<<dim3(TOK), dim3(128), 0, stream>>>(
            hres, norm_g + (long)i * DM_, norm_b + (long)i * DM_, hnb, DM_);
        gemm(hnb, w_in + (long)i * 2048 * 512, nullptr, nullptr, xz, nullptr,
             TOK, 2048, 512, 512, 0);
        conv_silu<<<dim3(TOK * DI_ / 256), dim3(256), 0, stream>>>(
            xz, conv_w + (long)i * DI_ * DC_, conv_b + (long)i * DI_, ucf, ucb);
        gemm(ucb, w_xprj + (long)i * 64 * 1024, nullptr, nullptr, dbl, dblb,
             TOK, 64, 1024, 1024, 0);
        gemm(dblb, w_dt + (long)i * 1024 * 32, dt_b + (long)i * DI_, nullptr,
             delta, nullptr, TOK, 1024, 32, 64, 2);                           // softplus
        mamba_scan<<<dim3(DI_ / 256, B_), dim3(256), 0, stream>>>(
            dbl, delta, ucf, A_log + (long)i * DI_ * NST_, ys);
        gate_out<<<dim3(TOK * DI_ / 256), dim3(256), 0, stream>>>(
            xz, ys, ucf, D_skip + (long)i * DI_, yb);
        gemm(yb, w_out + (long)i * 512 * 1024, nullptr, hres, hres, nullptr,
             TOK, 512, 1024, 1024, 0);                                        // residual add
    }

    // head
    layernorm_bf16<<<dim3(TOK), dim3(128), 0, stream>>>(hres, hn_g, hn_b, hnb, DM_);
    gemm(hnb, w_h1, h1_b, nullptr, nullptr, t1b, TOK, 128, 512, 512, 1);      // silu
    gemm(t1b, w_h2p, b_h2p, nullptr, logits, nullptr, TOK, HN2, 128, 128, 0);
    write_out<<<dim3(B_ * NC_ * T_ / 256), dim3(256), 0, stream>>>(logits, (float*)d_out);
}
